// MultiHeadAttention_867583393876
// MI455X (gfx1250) — matrix operand order repaired and checked by reading, not compiled, not run on hardware
//
#include <hip/hip_runtime.h>
#include <hip/hip_bf16.h>
#include <math.h>

// ---------------------------------------------------------------------------
// MHA with reference quirks, B=16,T=512,E=512,H=8.
//   scores = Q @ K     (reshape quirk: K memory reused as (d,s))
//   softmax over the QUERY axis (column softmax of the (t,s) matrix)
//   out    = reshape(Z,(b,512,4096)) @ WO   (reshape quirk == raw Z memory)
// Heavy math in bf16 WMMA (v_wmma_f32_16x16x32_bf16), f32 accumulate.
// A tiles stream through a DOUBLE-BUFFERED Tensor Data Mover pipeline:
// DMA(tile i+1) overlaps B-staging(i) + WMMA compute(i); s_wait_tensorcnt 1
// retires only the tile needed now. TDM padding reproduces the 40-element
// LDS row stride.
// ---------------------------------------------------------------------------

typedef __bf16 bf16_t;
typedef __attribute__((ext_vector_type(16))) __bf16    v16bf;
typedef __attribute__((ext_vector_type(8)))  __bf16    bf16x8;
typedef __attribute__((ext_vector_type(8)))  float     v8f;
typedef __attribute__((ext_vector_type(4)))  unsigned  v4u;
typedef __attribute__((ext_vector_type(8)))  unsigned  v8u;

#define LDS_STRIDE 40                 // 32 bf16 payload + 8 bf16 pad
#define A_BUF_ELTS (128 * LDS_STRIDE) // one A tile buffer (5120 bf16 = 10KB)

// ---------------------------------------------------------------- fp32->bf16
__global__ void mha_cvt_bf16(const float* __restrict__ src,
                             bf16_t* __restrict__ dst, int n) {
  int i = blockIdx.x * blockDim.x + threadIdx.x;
  int stride = gridDim.x * blockDim.x;
  for (; i < n; i += stride) dst[i] = (bf16_t)src[i];
}

// ------------------------------------------------- generic batched WMMA GEMM
// C[bz] = alpha * A[bz/aDiv] @ B[bz%bMod]
// A: MxK row-major bf16, B: KxN row-major bf16, C: MxN row-major (CT).
// Block tile 128x128, K-step 32, 8 wave32 per block, wave tile 32x64.
template <typename CT>
__global__ __launch_bounds__(256) void mha_gemm_bf16_wmma(
    const bf16_t* __restrict__ A, const bf16_t* __restrict__ B,
    CT* __restrict__ C, int M, int N, int K,
    long sA, int aDiv, long sB, int bMod, long sC, float alpha) {
  const int bz = blockIdx.z;
  A += sA * (long)(bz / aDiv);
  B += sB * (long)(bz % bMod);
  C += sC * (long)bz;
  const int lda = K, ldb = N, ldc = N;

  const int m0 = blockIdx.y * 128;
  const int n0 = blockIdx.x * 128;
  const int tid  = threadIdx.x;
  const int lane = tid & 31;
  const int w    = tid >> 5;
  const int wm   = (w >> 1) * 32;   // wave row offset in block tile
  const int wn   = (w & 1) * 64;    // wave col offset in block tile
  const int l16  = lane & 15;
  const int g    = lane >> 4;       // lane half-group

  __shared__ bf16_t lsA[2 * A_BUF_ELTS];   // [buf][row][k] row-major, TDM dest
  __shared__ bf16_t lsB[A_BUF_ELTS];       // [n][k]        TRANSPOSED

  v8f acc[2][4];
  {
    v8f z = {};
#pragma unroll
    for (int mf = 0; mf < 2; ++mf)
#pragma unroll
      for (int nf = 0; nf < 4; ++nf) acc[mf][nf] = z;
  }

  // ---- Tensor DMA descriptor group 1 (constant across K loop) ----
  // data_size=2B (code 1); pad_enable; pad_interval=3 (every 16 dwords);
  // pad_amount=3 (4 dwords)  -> reproduces LDS_STRIDE=40 bf16 rows.
  // tensor_dim0=K, tensor_dim1=512, tile=32x128, dim0_stride=K.
  const unsigned kdw = (unsigned)K;
  const v8u tdm_g1 = {
      (1u << 16) | (1u << 20) | (3u << 22) | (3u << 25),  // dsize|pad cfg
      (kdw << 16),                                        // tensor_dim0 lo
      (kdw >> 16) | (512u << 16),                         // dim0 hi | dim1 lo
      (32u << 16),                                        // dim1 hi | tile_dim0
      128u,                                               // tile_dim1 (| dim2=0)
      kdw,                                                // dim0_stride lo
      0u, 0u };                                           // stride hi / dim1_stride
  const unsigned lds_a_off = (unsigned)(size_t)&lsA[0];   // LDS byte offset

  // issue DMA of A tile kt (rows m0..m0+127, cols kt*32..+31) into buffer b
  auto tdm_issue = [&](int kt, int b) {
    unsigned long long ga =
        (unsigned long long)(size_t)(A + (long)m0 * lda + kt * 32);
    v4u g0 = { 1u,                                     // count=1, no gather
               lds_a_off + (unsigned)b * (A_BUF_ELTS * 2u),  // lds_addr
               (unsigned)ga,                           // global_addr[31:0]
               (unsigned)((ga >> 32) & 0x01FFFFFFu) | 0x80000000u }; // type=2
    asm volatile("tensor_load_to_lds %0, %1" :: "s"(g0), "s"(tdm_g1)
                 : "memory");
  };

  const int nk = K >> 5;           // number of 32-wide K tiles
  if (w == 0) tdm_issue(0, 0);     // prologue: DMA tile 0 -> buffer 0

  for (int kt = 0; kt < nk; ++kt) {
    const int k0  = kt * 32;
    const bf16_t* curA = &lsA[(kt & 1) * A_BUF_ELTS];

    // ---- overlap: DMA NEXT A tile into the alternate buffer ----
    if (kt + 1 < nk && w == 0) tdm_issue(kt + 1, (kt + 1) & 1);

    // ---- stage B tile transposed: read coalesced rows, scatter to [n][k] ----
#pragma unroll
    for (int i = 0; i < 2; ++i) {
      int c  = tid + 256 * i;            // 0..511
      int kr = c >> 4, cc = c & 15;
      bf16x8 v = *(const bf16x8*)&B[(long)(k0 + kr) * ldb + n0 + cc * 8];
#pragma unroll
      for (int j = 0; j < 8; ++j)
        lsB[(cc * 8 + j) * LDS_STRIDE + kr] = v[j];
    }
    if (kt + 1 < nk) {   // hint next B tile toward L2 (global_prefetch_b8)
      __builtin_prefetch(&B[(long)(k0 + 32 + (tid >> 3)) * ldb + n0 + (tid & 7) * 16], 0, 0);
    }

    // ---- retire the DMA for THIS tile (next one may stay in flight) ----
    if (kt + 1 < nk) __builtin_amdgcn_s_wait_tensorcnt(1);
    else             __builtin_amdgcn_s_wait_tensorcnt(0);
    __syncthreads();

    // ---- A fragments (ISA 16-bit A 16x32 layout):
    // lane half g: VGPRs0-3 <- K = g*8+0..7, VGPRs4-7 <- K = 16+g*8+0..7
    v16bf af[2];
#pragma unroll
    for (int mf = 0; mf < 2; ++mf) {
      int row   = wm + mf * 16 + l16;
      bf16x8 lo = *(const bf16x8*)&curA[row * LDS_STRIDE + g * 8];
      bf16x8 hi = *(const bf16x8*)&curA[row * LDS_STRIDE + 16 + g * 8];
      af[mf] = __builtin_shufflevector(lo, hi, 0, 1, 2, 3, 4, 5, 6, 7,
                                       8, 9, 10, 11, 12, 13, 14, 15);
    }
    // ---- B fragments: lane holds column n=l16, rows K = g*8+0..7, 16+g*8+0..7
#pragma unroll
    for (int nf = 0; nf < 4; ++nf) {
      int col   = wn + nf * 16 + l16;
      bf16x8 lo = *(const bf16x8*)&lsB[col * LDS_STRIDE + g * 8];
      bf16x8 hi = *(const bf16x8*)&lsB[col * LDS_STRIDE + 16 + g * 8];
      v16bf bfr = __builtin_shufflevector(lo, hi, 0, 1, 2, 3, 4, 5, 6, 7,
                                          8, 9, 10, 11, 12, 13, 14, 15);
#pragma unroll
      for (int mf = 0; mf < 2; ++mf)
        acc[mf][nf] = __builtin_amdgcn_wmma_f32_16x16x32_bf16(
            false, af[mf], false, bfr, (short)0, acc[mf][nf], false, false);
    }
    __syncthreads();   // lsB (and the drained A buffer) are free again
  }

  // ---- epilogue: D layout = lane l16 is N, VGPR r -> M = g*8 + r ----
#pragma unroll
  for (int mf = 0; mf < 2; ++mf)
#pragma unroll
    for (int nf = 0; nf < 4; ++nf) {
      int colc = n0 + wn + nf * 16 + l16;
#pragma unroll
      for (int r = 0; r < 8; ++r) {
        int rowc = m0 + wm + mf * 16 + g * 8 + r;
        C[(long)rowc * ldc + colc] = (CT)(acc[mf][nf][r] * alpha);
      }
    }
}

// ------------------------------- column softmax stats (softmax over t axis!)
// One block per (b,h). Per column s: online max/sum over t of S[t,s].
__global__ __launch_bounds__(256) void mha_colstats(
    const bf16_t* __restrict__ S, float* __restrict__ stats, int T) {
  const bf16_t* Sb = S + (long)blockIdx.x * T * T;
  float* st = stats + (long)blockIdx.x * 2 * T;
  for (int s = threadIdx.x; s < T; s += blockDim.x) {
    float m = -3.0e38f, l = 0.0f;
    for (int t = 0; t < T; ++t) {
      float v  = (float)Sb[(long)t * T + s];
      float nm = fmaxf(m, v);
      l = l * __expf(m - nm) + __expf(v - nm);
      m = nm;
    }
    st[s]     = m;
    st[T + s] = l;
  }
}

// ------------------------- in-place P[t,s] = exp(S[t,s]-max[s]) / sum[s]
__global__ __launch_bounds__(256) void mha_expnorm(
    bf16_t* __restrict__ S, const float* __restrict__ stats, int T, long total) {
  long i      = (long)blockIdx.x * blockDim.x + threadIdx.x;
  long stride = (long)gridDim.x * blockDim.x;
  long per    = (long)T * T;
  for (; i < total; i += stride) {
    long bh = i / per;
    int  s  = (int)(i % (long)T);
    const float* st = stats + bh * 2 * T;
    float v = (float)S[i];
    S[i] = (bf16_t)(__expf(v - st[s]) / st[T + s]);
  }
}

// ---------------------------------------------------------------------------
extern "C" void kernel_launch(void* const* d_in, const int* in_sizes, int n_in,
                              void* d_out, int out_size, void* d_ws, size_t ws_size,
                              hipStream_t stream) {
  (void)in_sizes; (void)n_in; (void)out_size; (void)ws_size;
  constexpr int Bb = 16, T = 512, E = 512, H = 8;
  constexpr long ME  = (long)T * E;       // 262144 per (b or h) matrix
  constexpr long QKV = (long)Bb * H * ME; // 33554432

  const float* x  = (const float*)d_in[0];
  const float* WQ = (const float*)d_in[1];
  const float* WK = (const float*)d_in[2];
  const float* WV = (const float*)d_in[3];
  const float* WO = (const float*)d_in[4];
  float* out = (float*)d_out;

  char* ws = (char*)d_ws;
  auto alloc = [&](size_t bytes) -> void* {
    void* p = ws;
    ws += (bytes + 255) & ~(size_t)255;
    return p;
  };
  bf16_t* xb  = (bf16_t*)alloc((size_t)Bb * ME * 2);
  bf16_t* wqb = (bf16_t*)alloc((size_t)H * ME * 2);
  bf16_t* wkb = (bf16_t*)alloc((size_t)H * ME * 2);
  bf16_t* wvb = (bf16_t*)alloc((size_t)H * ME * 2);
  bf16_t* wob = (bf16_t*)alloc((size_t)H * ME * 2);
  bf16_t* Qb  = (bf16_t*)alloc((size_t)QKV * 2);
  bf16_t* Kb  = (bf16_t*)alloc((size_t)QKV * 2);
  bf16_t* Vb  = (bf16_t*)alloc((size_t)QKV * 2);
  bf16_t* Sb  = (bf16_t*)alloc((size_t)QKV * 2);   // scores, then P in place
  bf16_t* Zb  = (bf16_t*)alloc((size_t)QKV * 2);
  float*  stats = (float*)alloc((size_t)Bb * H * 2 * T * 4);

  // fp32 -> bf16 conversions
  mha_cvt_bf16<<<4096, 256, 0, stream>>>(x,  xb,  (int)(Bb * ME));
  mha_cvt_bf16<<<2048, 256, 0, stream>>>(WQ, wqb, (int)(H * ME));
  mha_cvt_bf16<<<2048, 256, 0, stream>>>(WK, wkb, (int)(H * ME));
  mha_cvt_bf16<<<2048, 256, 0, stream>>>(WV, wvb, (int)(H * ME));
  mha_cvt_bf16<<<2048, 256, 0, stream>>>(WO, wob, (int)(H * ME));

  dim3 blk(256);
  dim3 gBH(512 / 128, 512 / 128, Bb * H);   // (4,4,128)

  // Q/K/V: out[b,h] = x[b] @ W[h]
  mha_gemm_bf16_wmma<bf16_t><<<gBH, blk, 0, stream>>>(
      xb, wqb, Qb, 512, 512, 512, ME, H, ME, H, ME, 1.0f);
  mha_gemm_bf16_wmma<bf16_t><<<gBH, blk, 0, stream>>>(
      xb, wkb, Kb, 512, 512, 512, ME, H, ME, H, ME, 1.0f);
  mha_gemm_bf16_wmma<bf16_t><<<gBH, blk, 0, stream>>>(
      xb, wvb, Vb, 512, 512, 512, ME, H, ME, H, ME, 1.0f);

  // scores = (Q @ K) / sqrt(512)   (reshape quirk: K used as-is)
  const float inv_sqrt_t = 0.044194173824159216f;   // 1/sqrt(512)
  mha_gemm_bf16_wmma<bf16_t><<<gBH, blk, 0, stream>>>(
      Qb, Kb, Sb, 512, 512, 512, ME, 1, ME, Bb * H, ME, inv_sqrt_t);

  // softmax over the query axis (column softmax)
  mha_colstats<<<Bb * H, 256, 0, stream>>>(Sb, stats, T);
  mha_expnorm<<<8192, 256, 0, stream>>>(Sb, stats, T, QKV);

  // Z = P @ V
  mha_gemm_bf16_wmma<bf16_t><<<gBH, blk, 0, stream>>>(
      Sb, Vb, Zb, 512, 512, 512, ME, 1, ME, Bb * H, ME, 1.0f);

  // out[b] = reshape(Z[b],(512,4096)) @ WO   (reshape quirk == raw Z memory)
  dim3 gOut(512 / 128, 512 / 128, Bb);      // (4,4,16)
  mha_gemm_bf16_wmma<float><<<gOut, blk, 0, stream>>>(
      Zb, wob, out, 512, 512, 4096, (long)H * ME, 1, 0L, 1, ME, 1.0f);
}
